// Net_32847909880071
// MI455X (gfx1250) — matrix operand order repaired and checked by reading, not compiled, not run on hardware
//
#include <hip/hip_runtime.h>
#include <hip/hip_bf16.h>
#include <math.h>

// ---------------------------------------------------------------------------
// Types
// ---------------------------------------------------------------------------
typedef __attribute__((ext_vector_type(16))) __bf16        v16bf;
typedef __attribute__((ext_vector_type(8)))  float         v8f;
typedef __attribute__((ext_vector_type(4)))  unsigned int  v4u;
typedef __attribute__((ext_vector_type(4)))  unsigned short v4us;
typedef unsigned short us;

union FragBF {            // one wave-striped 16x32 (A) or 32x16 (B) bf16 fragment
  v16bf v;
  v4u   q[2];
  us    u[16];
};

static __device__ __forceinline__ us f32_to_bf16_rne(float f) {
  unsigned int u = __float_as_uint(f);
  u += 0x7FFFu + ((u >> 16) & 1u);      // round-to-nearest-even
  return (us)(u >> 16);
}

// ---------------------------------------------------------------------------
// Elementwise / conversion kernels
// ---------------------------------------------------------------------------
__global__ void k_fill(float* __restrict__ a, float v, int n) {
  int i = blockIdx.x * blockDim.x + threadIdx.x;
  if (i < n) a[i] = v;
}

__global__ void k_deg_edges(const int* __restrict__ ei, int E, float* __restrict__ deg) {
  int e = blockIdx.x * blockDim.x + threadIdx.x;
  if (e < E) atomicAdd(&deg[ei[E + e]], 1.0f);
}

__global__ void k_rsqrt(float* __restrict__ a, int n) {
  int i = blockIdx.x * blockDim.x + threadIdx.x;
  if (i < n) a[i] = rsqrtf(a[i]);       // deg >= 1 always (self-loops)
}

// float4 -> 4x bf16 (optionally fused ReLU); n4 = n/4 elements
template <int RELU>
__global__ void k_f32_to_bf16_v4(const float* __restrict__ in, us* __restrict__ out,
                                 long long n4) {
  long long i = (long long)blockIdx.x * blockDim.x + threadIdx.x;
  if (i >= n4) return;
  float4 f = ((const float4*)in)[i];
  if (RELU) {
    f.x = fmaxf(f.x, 0.0f); f.y = fmaxf(f.y, 0.0f);
    f.z = fmaxf(f.z, 0.0f); f.w = fmaxf(f.w, 0.0f);
  }
  v4us o;
  o[0] = f32_to_bf16_rne(f.x);
  o[1] = f32_to_bf16_rne(f.y);
  o[2] = f32_to_bf16_rne(f.z);
  o[3] = f32_to_bf16_rne(f.w);
  ((v4us*)out)[i] = o;
}

// weight convert + transpose:  Wt[n][k] = bf16(W[k][n]);  W is [K][N] row-major
__global__ void k_w_transpose_bf16(const float* __restrict__ W, us* __restrict__ Wt,
                                   int K, int N) {
  int t = blockIdx.x * blockDim.x + threadIdx.x;
  if (t >= K * N) return;
  int k = t / N, n = t % N;               // coalesced read along n
  Wt[(size_t)n * K + k] = f32_to_bf16_rne(W[t]);
}

// ---------------------------------------------------------------------------
// GCN aggregation: out[col] = sum_e dis[row]*dis[col]*h[row]  (+ self loop + bias)
// ---------------------------------------------------------------------------
__global__ void k_agg_init(const float* __restrict__ h, const float* __restrict__ dis,
                           const float* __restrict__ bias, float* __restrict__ out,
                           int n, int D, int ldo, int colOff) {
  long long t = (long long)blockIdx.x * blockDim.x + threadIdx.x;
  int Q = D >> 2;
  if (t >= (long long)n * Q) return;
  int i = (int)(t / Q), q = (int)(t % Q);
  float w = dis[i] * dis[i];              // self-loop norm
  float4 hv = *(const float4*)(h + (size_t)i * D + q * 4);
  float4 bv = *(const float4*)(bias + q * 4);
  float4 o = make_float4(w * hv.x + bv.x, w * hv.y + bv.y,
                         w * hv.z + bv.z, w * hv.w + bv.w);
  *(float4*)(out + (size_t)i * ldo + colOff + q * 4) = o;
}

__global__ void k_agg_edges(const float* __restrict__ h, const float* __restrict__ dis,
                            const int* __restrict__ ei, int E, int D, int ldo, int colOff,
                            float* __restrict__ out) {
  long long t = (long long)blockIdx.x * blockDim.x + threadIdx.x;
  int Q = D >> 2;
  if (t >= (long long)E * Q) return;
  int e = (int)(t / Q), q = (int)(t % Q);
  int row = ei[e], col = ei[E + e];
  float w = dis[row] * dis[col];
  float4 hv = *(const float4*)(h + (size_t)row * D + q * 4);
  float* dst = out + (size_t)col * ldo + colOff + q * 4;
  atomicAdd(dst + 0, w * hv.x);
  atomicAdd(dst + 1, w * hv.y);
  atomicAdd(dst + 2, w * hv.z);
  atomicAdd(dst + 3, w * hv.w);
}

// ---------------------------------------------------------------------------
// Segment-mean pooling
// ---------------------------------------------------------------------------
__global__ void k_count(const int* __restrict__ idx, int n, float* __restrict__ cnt) {
  int i = blockIdx.x * blockDim.x + threadIdx.x;
  if (i < n) atomicAdd(&cnt[idx[i]], 1.0f);
}

__global__ void k_pool_sum(const float* __restrict__ h, const int* __restrict__ idx,
                           int n, int D, int ldo, int colOff, float* __restrict__ out) {
  long long t = (long long)blockIdx.x * blockDim.x + threadIdx.x;
  int Q = D >> 2;
  if (t >= (long long)n * Q) return;
  int i = (int)(t / Q), q = (int)(t % Q);
  int s = idx[i];
  float4 hv = *(const float4*)(h + (size_t)i * D + q * 4);
  float* dst = out + (size_t)s * ldo + colOff + q * 4;
  atomicAdd(dst + 0, hv.x);
  atomicAdd(dst + 1, hv.y);
  atomicAdd(dst + 2, hv.z);
  atomicAdd(dst + 3, hv.w);
}

__global__ void k_pool_div(float* __restrict__ P, const float* __restrict__ c1,
                           const float* __restrict__ c2, int S, int D2, int D) {
  long long t = (long long)blockIdx.x * blockDim.x + threadIdx.x;
  if (t >= (long long)S * D2) return;
  int s = (int)(t / D2), c = (int)(t % D2);
  float cn = (c < D) ? c1[s] : c2[s];
  P[t] = P[t] / fmaxf(cn, 1.0f);
}

// ---------------------------------------------------------------------------
// Final 256->7 linear + log_softmax (tiny; one thread per segment)
// ---------------------------------------------------------------------------
__global__ void k_head(const float* __restrict__ PH, const float* __restrict__ W2,
                       const float* __restrict__ b2, float* __restrict__ out,
                       int S, int D, int C) {
  int s = blockIdx.x * blockDim.x + threadIdx.x;
  if (s >= S) return;
  float logit[16];
  for (int c = 0; c < C; ++c) logit[c] = b2[c];
  const float* hp = PH + (size_t)s * D;
  for (int k = 0; k < D; ++k) {
    float hv = hp[k];
    const float* wr = W2 + (size_t)k * C;
    for (int c = 0; c < C; ++c) logit[c] += hv * wr[c];
  }
  float m = logit[0];
  for (int c = 1; c < C; ++c) m = fmaxf(m, logit[c]);
  float sum = 0.0f;
  for (int c = 0; c < C; ++c) sum += expf(logit[c] - m);
  float lse = m + logf(sum);
  for (int c = 0; c < C; ++c) out[(size_t)s * C + c] = logit[c] - lse;
}

// ---------------------------------------------------------------------------
// WMMA bf16 GEMM:  C[M,N] = A[M,K](bf16) @ Bt[N,K](bf16, pre-transposed weights)
// Block tile 128x128, 256 threads = 8 waves, each wave 32x64 (2x4 WMMA tiles),
// K stepped by 32; double-buffered LDS (ping-pong) so the next tile's global
// loads overlap the current tile's v_wmma stream.
// Requires: K % 32 == 0, N % 128 == 0, 16B-aligned A/Bt.
// ---------------------------------------------------------------------------
template <int HASBIAS, int RELU>
__global__ __launch_bounds__(256) void k_gemm_bf16(
    const us* __restrict__ A, const us* __restrict__ Bt,
    const float* __restrict__ bias, float* __restrict__ Cout,
    int M, int N, int K, int ldc) {
  __shared__ __align__(16) us As[2][128 * 32];   // [row][k]
  __shared__ __align__(16) us Bs[2][128 * 32];   // [n][k]

  const int tid   = threadIdx.x;
  const int lane  = tid & 31;
  const int wid   = tid >> 5;          // 0..7
  const int waveM = wid >> 1;          // 0..3  (rows, *32)
  const int waveN = wid & 1;           // 0..1  (cols, *64)
  const int hi    = lane >> 4;         // lane-group (0/1)
  const int l15   = lane & 15;

  const int blockN0 = blockIdx.x * 128;
  const int blockM0 = blockIdx.y * 128;

  // global->LDS staging mapping (identical for A and Bt)
  const int ar   = tid >> 1;           // tile row (A: m, B: n), 0..127
  const int ac   = (tid & 1) * 16;     // k chunk (0 or 16)
  const int grow = blockM0 + ar;       // global A row
  const int gcol = blockN0 + ar;       // global Bt row (output column)

  const v4u vz = {};
  v8f zacc = {};
  v8f acc[2][4];
#pragma unroll
  for (int i = 0; i < 2; ++i)
#pragma unroll
    for (int j = 0; j < 4; ++j) acc[i][j] = zacc;

  v4u a0, a1, b0, b1;

  // ---- prologue: stage tile 0 into buffer 0 ----
  a0 = vz; a1 = vz;
  if (grow < M) {
    const us* s = A + (size_t)grow * K + ac;
    a0 = *(const v4u*)(s);
    a1 = *(const v4u*)(s + 8);
  }
  {
    const us* s = Bt + (size_t)gcol * K + ac;
    b0 = *(const v4u*)(s);
    b1 = *(const v4u*)(s + 8);
  }
  *(v4u*)(&As[0][ar * 32 + ac])     = a0;
  *(v4u*)(&As[0][ar * 32 + ac + 8]) = a1;
  *(v4u*)(&Bs[0][ar * 32 + ac])     = b0;
  *(v4u*)(&Bs[0][ar * 32 + ac + 8]) = b1;
  __syncthreads();

  const int KT = K >> 5;
  for (int t = 0; t < KT; ++t) {
    const int cur  = t & 1;
    const int nxt  = cur ^ 1;
    const bool more = (t + 1 < KT);

    // ---- issue next tile's global loads (overlap with WMMA below) ----
    if (more) {
      const int k0 = (t + 1) << 5;
      a0 = vz; a1 = vz;
      if (grow < M) {
        const us* s = A + (size_t)grow * K + k0 + ac;
        a0 = *(const v4u*)(s);
        a1 = *(const v4u*)(s + 8);
        if (t + 2 < KT) __builtin_prefetch(s + 32, 0, 1);   // global_prefetch_b8
      }
      const us* s = Bt + (size_t)gcol * K + k0 + ac;
      b0 = *(const v4u*)(s);
      b1 = *(const v4u*)(s + 8);
    }

    // ---- build fragments (documented 16-bit A/B wave32 layouts) ----
    FragBF afrag[2], bfrag[4];
#pragma unroll
    for (int im = 0; im < 2; ++im) {
      // A 16x32: lane l15 = row; elems 0..7 -> K = hi*8 + i, elems 8..15 -> 16 + hi*8 + i
      const us* s = &As[cur][(waveM * 32 + im * 16 + l15) * 32 + hi * 8];
      afrag[im].q[0] = *(const v4u*)(s);
      afrag[im].q[1] = *(const v4u*)(s + 16);
    }
#pragma unroll
    for (int in = 0; in < 4; ++in) {
      // B 32x16: lane l15 = col; elems 0..7 -> K = hi*8 + i, elems 8..15 -> 16 + hi*8 + i
      const us* s = &Bs[cur][(waveN * 64 + in * 16 + l15) * 32 + hi * 8];
      bfrag[in].q[0] = *(const v4u*)(s);
      bfrag[in].q[1] = *(const v4u*)(s + 16);
    }

    // ---- 8 WMMAs per wave per K-step ----
#pragma unroll
    for (int im = 0; im < 2; ++im)
#pragma unroll
      for (int in = 0; in < 4; ++in)
        acc[im][in] = __builtin_amdgcn_wmma_f32_16x16x32_bf16(
            false, afrag[im].v, false, bfrag[in].v,
            (short)0, acc[im][in], false, false);

    // ---- commit next tile to alternate LDS buffer ----
    if (more) {
      *(v4u*)(&As[nxt][ar * 32 + ac])     = a0;
      *(v4u*)(&As[nxt][ar * 32 + ac + 8]) = a1;
      *(v4u*)(&Bs[nxt][ar * 32 + ac])     = b0;
      *(v4u*)(&Bs[nxt][ar * 32 + ac + 8]) = b1;
    }
    __syncthreads();
  }

  // ---- epilogue: C layout = lane l15 -> col, VGPR r -> row r + hi*8 ----
#pragma unroll
  for (int im = 0; im < 2; ++im) {
#pragma unroll
    for (int in = 0; in < 4; ++in) {
      int col = blockN0 + waveN * 64 + in * 16 + l15;
      float bv = 0.0f;
      if (HASBIAS) bv = bias[col];
#pragma unroll
      for (int r = 0; r < 8; ++r) {
        int row = blockM0 + waveM * 32 + im * 16 + hi * 8 + r;
        if (row < M) {
          float v = acc[im][in][r] + bv;
          if (RELU) v = fmaxf(v, 0.0f);
          Cout[(size_t)row * ldc + col] = v;
        }
      }
    }
  }
}

// ---------------------------------------------------------------------------
// Host orchestration
// ---------------------------------------------------------------------------
static inline int n_blocks(long long n, int t) { return (int)((n + t - 1) / t); }

extern "C" void kernel_launch(void* const* d_in, const int* in_sizes, int n_in,
                              void* d_out, int out_size, void* d_ws, size_t ws_size,
                              hipStream_t stream) {
  (void)n_in; (void)ws_size;
  const float* x    = (const float*)d_in[0];
  const int*   ei1  = (const int*)d_in[1];
  const int*   ei2  = (const int*)d_in[2];
  const int*   idx1 = (const int*)d_in[3];
  const int*   idx2 = (const int*)d_in[4];
  const float* w11  = (const float*)d_in[5];  const float* b11  = (const float*)d_in[6];
  const float* w12  = (const float*)d_in[7];  const float* b12  = (const float*)d_in[8];
  const float* w21  = (const float*)d_in[9];  const float* b21  = (const float*)d_in[10];
  const float* w22  = (const float*)d_in[11]; const float* b22  = (const float*)d_in[12];
  const float* m1w1 = (const float*)d_in[13]; const float* m1b1 = (const float*)d_in[14];
  const float* m1w2 = (const float*)d_in[15]; const float* m1b2 = (const float*)d_in[16];
  const float* m2w1 = (const float*)d_in[17]; const float* m2b1 = (const float*)d_in[18];
  const float* m2w2 = (const float*)d_in[19]; const float* m2b2 = (const float*)d_in[20];
  const float* mw1  = (const float*)d_in[21]; const float* mb1  = (const float*)d_in[22];
  const float* mw2  = (const float*)d_in[23]; const float* mb2  = (const float*)d_in[24];

  const int N  = in_sizes[3];          // 100000 nodes
  const int E  = in_sizes[1] / 2;      // 800000 edges
  const int F  = in_sizes[0] / N;      // 128
  const int D  = in_sizes[6];          // 256
  const int C  = in_sizes[24];         // 7
  const int S  = out_size / C;         // 5000
  const int D2 = 2 * D;

  // ---- carve workspace (all offsets 256B aligned) ----
  char* base = (char*)d_ws; size_t off = 0;
  auto carve = [&](size_t bytes) {
    void* r = base + off;
    off += bytes; off = (off + 255) & ~(size_t)255;
    return r;
  };
  float* dis1 = (float*)carve(sizeof(float) * (size_t)N);
  float* dis2 = (float*)carve(sizeof(float) * (size_t)N);
  float* HW   = (float*)carve(sizeof(float) * (size_t)N * D);   // GEMM out / mlp hidden
  float* HCAT = (float*)carve(sizeof(float) * (size_t)N * D2);  // [x1|x2] pre-relu
  float* H    = HCAT;  // layer output aliases HCAT (HCAT is dead when H is written)
  us*    ABF  = (us*)carve(sizeof(us) * (size_t)N * D2);
  us*    WBT  = (us*)carve(sizeof(us) * (size_t)D2 * D);        // transposed bf16 weights
  float* P    = (float*)carve(sizeof(float) * (size_t)S * D2);
  float* PH   = (float*)carve(sizeof(float) * (size_t)S * D);
  float* CNT  = (float*)carve(sizeof(float) * (size_t)2 * S);

  const int T = 256;

  auto to_bf16 = [&](const float* src, us* dst, long long n, bool relu) {
    long long n4 = n >> 2;
    if (relu)
      k_f32_to_bf16_v4<1><<<n_blocks(n4, T), T, 0, stream>>>(src, dst, n4);
    else
      k_f32_to_bf16_v4<0><<<n_blocks(n4, T), T, 0, stream>>>(src, dst, n4);
  };

  auto gemm = [&](const us* Abf, const float* Wf, int K, const float* bias,
                  bool relu, float* Cout, int M) {
    k_w_transpose_bf16<<<n_blocks((long long)K * D, T), T, 0, stream>>>(Wf, WBT, K, D);
    dim3 g((unsigned)(D / 128), (unsigned)n_blocks(M, 128), 1);
    if (bias && relu)
      k_gemm_bf16<1, 1><<<g, T, 0, stream>>>(Abf, WBT, bias, Cout, M, D, K, D);
    else if (bias)
      k_gemm_bf16<1, 0><<<g, T, 0, stream>>>(Abf, WBT, bias, Cout, M, D, K, D);
    else
      k_gemm_bf16<0, 0><<<g, T, 0, stream>>>(Abf, WBT, nullptr, Cout, M, D, K, D);
  };

  auto gcn = [&](int K, const float* Wf, const float* bias, const int* ei,
                 const float* dis, int colOff) {
    gemm(ABF, Wf, K, nullptr, false, HW, N);  // h = X @ W (no bias yet)
    k_agg_init<<<n_blocks((long long)N * (D / 4), T), T, 0, stream>>>(
        HW, dis, bias, HCAT, N, D, D2, colOff);
    k_agg_edges<<<n_blocks((long long)E * (D / 4), T), T, 0, stream>>>(
        HW, dis, ei, E, D, D2, colOff, HCAT);
  };

  // ---- degree -> deg^{-1/2} per edge list (self-loop counted via init=1) ----
  k_fill<<<n_blocks(N, T), T, 0, stream>>>(dis1, 1.0f, N);
  k_fill<<<n_blocks(N, T), T, 0, stream>>>(dis2, 1.0f, N);
  k_deg_edges<<<n_blocks(E, T), T, 0, stream>>>(ei1, E, dis1);
  k_deg_edges<<<n_blocks(E, T), T, 0, stream>>>(ei2, E, dis2);
  k_rsqrt<<<n_blocks(N, T), T, 0, stream>>>(dis1, N);
  k_rsqrt<<<n_blocks(N, T), T, 0, stream>>>(dis2, N);

  // ---- layer 1 ----
  to_bf16(x, ABF, (long long)N * F, false);
  gcn(F, w11, b11, ei1, dis1, 0);
  gcn(F, w12, b12, ei2, dis2, D);
  to_bf16(HCAT, ABF, (long long)N * D2, true);           // fused ReLU(x1|x2) -> bf16
  gemm(ABF, m1w1, D2, m1b1, true, HW, N);                // relu(cat @ m1w1 + b)
  to_bf16(HW, ABF, (long long)N * D, false);
  gemm(ABF, m1w2, D, m1b2, false, H, N);                 // h = hidden @ m1w2 + b

  // ---- layer 2 ----
  to_bf16(H, ABF, (long long)N * D, false);
  gcn(D, w21, b21, ei1, dis1, 0);
  gcn(D, w22, b22, ei2, dis2, D);
  to_bf16(HCAT, ABF, (long long)N * D2, true);
  gemm(ABF, m2w1, D2, m2b1, true, HW, N);
  to_bf16(HW, ABF, (long long)N * D, false);
  gemm(ABF, m2w2, D, m2b2, false, H, N);

  // ---- scatter-mean pooling (both index sets into P = [p1|p2]) ----
  hipMemsetAsync(P, 0, sizeof(float) * (size_t)S * D2, stream);
  hipMemsetAsync(CNT, 0, sizeof(float) * (size_t)2 * S, stream);
  k_count<<<n_blocks(N, T), T, 0, stream>>>(idx1, N, CNT);
  k_count<<<n_blocks(N, T), T, 0, stream>>>(idx2, N, CNT + S);
  k_pool_sum<<<n_blocks((long long)N * (D / 4), T), T, 0, stream>>>(H, idx1, N, D, D2, 0, P);
  k_pool_sum<<<n_blocks((long long)N * (D / 4), T), T, 0, stream>>>(H, idx2, N, D, D2, D, P);
  k_pool_div<<<n_blocks((long long)S * D2, T), T, 0, stream>>>(P, CNT, CNT + S, S, D2, D);

  // ---- head: relu(P @ mw1 + mb1) @ mw2 + mb2, log_softmax ----
  to_bf16(P, ABF, (long long)S * D2, false);
  gemm(ABF, mw1, D2, mb1, true, PH, S);
  k_head<<<n_blocks(S, T), T, 0, stream>>>(PH, mw2, mb2, (float*)d_out, S, D, C);
}
